// MockAWQLlamaLayer_39874476376633
// MI455X (gfx1250) — matrix operand order repaired and checked by reading, not compiled, not run on hardware
//
#include <hip/hip_runtime.h>
#include <hip/hip_bf16.h>
#include <math.h>

// ---------------------------------------------------------------------------
// MockAWQLlamaLayer for MI455X (gfx1250, wave32, WMMA bf16 path)
//   tokens M = 8*16 = 128, H = 4096, I = 11008, AWQ group G = 128
// Activations are kept in bf16 between ops. GEMM A-tiles are staged with
// global_load_async_to_lds_b128 (ASYNCcnt); weights are dequantized in
// registers and stored transposed to LDS. 8 v_wmma per wave per k-tile.
// ---------------------------------------------------------------------------

#define TOK   128
#define HDIM  4096
#define IDIM  11008
#define KTILE 64
#define NTILE 64
#define LSTR  72   // LDS row stride in halfs (64 + 8 pad, keeps 16B alignment)

typedef __attribute__((ext_vector_type(16))) __bf16 v16bf;
typedef __attribute__((ext_vector_type(8)))  float  v8f;

// fp32 -> bf16 (round-to-nearest-away; ties differ from RNE only)
__device__ __forceinline__ unsigned short bf16_rna(float f) {
  union { float f; unsigned u; } c; c.f = f;
  return (unsigned short)((c.u + 0x8000u) >> 16);
}

// two fp32 -> packed bf16x2 in one v_perm_b32
__device__ __forceinline__ unsigned pack_bf16x2(float a, float b) {
  union { float f; unsigned u; } x, y; x.f = a; y.f = b;
#if __has_builtin(__builtin_amdgcn_perm)
  // result bytes {0,1}=a bytes {2,3}; {2,3}=b bytes {2,3}
  return __builtin_amdgcn_perm(y.u + 0x8000u, x.u + 0x8000u, 0x07060302u);
#else
  return ((x.u + 0x8000u) >> 16) | ((y.u + 0x8000u) & 0xFFFF0000u);
#endif
}

__device__ __forceinline__ void wait_async0() {
  asm volatile("s_wait_asynccnt 0" ::: "memory");
}

struct StageRegs {
  int   qv0, qv1;  // packed AWQ weights for adjacent K rows (same nibble col)
  int   zv;        // packed zeros (shared: pair stays inside one G=128 group)
  float scv[8];    // scales for the 8 dequantized columns
};

// weight-tile global loads into registers (B side only; A goes async to LDS)
__device__ __forceinline__ void load_global_tileB(
    StageRegs& R, int tid, const int* __restrict__ qw,
    const int* __restrict__ qz, const float* __restrict__ sc,
    int Ntot, int qcols, int n0, int kk)
{
  const int kp   = tid >> 3;         // 0..31 k-pair
  const int cl   = tid & 7;          // 0..7  packed col
  const int krow = kk + (kp << 1);   // even local k
  const int kg   = krow >> 7;        // AWQ group (G=128)
  R.qv0 = qw[(size_t)krow       * qcols + (n0 >> 3) + cl];
  R.qv1 = qw[(size_t)(krow + 1) * qcols + (n0 >> 3) + cl];
  R.zv  = qz[(size_t)kg         * qcols + (n0 >> 3) + cl];
  const float* scp = &sc[(size_t)kg * Ntot + n0 + (cl << 3)];
  #pragma unroll
  for (int j = 0; j < 8; ++j) R.scv[j] = scp[j];
}

// dequant + store weight tile transposed [n][k] into LDS
__device__ __forceinline__ void store_lds_tileB(
    const StageRegs& R, int tid, unsigned short (*__restrict__ Bb)[LSTR])
{
  const int kp = tid >> 3;
  const int cl = tid & 7;
  const int k0 = kp << 1;
  const int sh[8] = {0, 16, 4, 20, 8, 24, 12, 28};   // AWQ nibble order
  #pragma unroll
  for (int j = 0; j < 8; ++j) {
    int   zn = (R.zv >> sh[j]) & 15;
    float f0 = (float)(((R.qv0 >> sh[j]) & 15) - zn) * R.scv[j];
    float f1 = (float)(((R.qv1 >> sh[j]) & 15) - zn) * R.scv[j];
    *(unsigned*)(&Bb[(cl << 3) + j][k0]) = pack_bf16x2(f0, f1);  // ds_store_b32
  }
}

// async-DMA the bf16 A tile (128 x 64 halfs) straight into LDS, no VGPR data
__device__ __forceinline__ void issue_async_A(
    int tid, const unsigned short* __restrict__ Abf, int K, int kk,
    unsigned short (*__restrict__ Ab)[LSTR])
{
  #pragma unroll
  for (int i = 0; i < 4; ++i) {
    int q  = i * 256 + tid;          // 0..1023 16-byte chunks
    int r  = q >> 3;                 // row 0..127 (8 chunks per row)
    int c8 = (q & 7) << 3;           // half offset 0,8,...,56
    const unsigned short* g = &Abf[(size_t)r * K + kk + c8];
    unsigned laddr = (unsigned)(size_t)&Ab[r][c8];   // 32-bit LDS offset
    asm volatile("global_load_async_to_lds_b128 %0, %1, off"
                 :: "v"(laddr), "v"(g) : "memory");
  }
}

// ---------------------------------------------------------------------------
template <bool CBF16>
__global__ __launch_bounds__(256, 1)
void gemm_awq_bf16(const unsigned short* __restrict__ A,  // 128 x K bf16
                   const int*   __restrict__ qw,   // K x (Ntot/8)
                   const int*   __restrict__ qz,   // (K/128) x (Ntot/8)
                   const float* __restrict__ sc,   // (K/128) x Ntot
                   void*        __restrict__ Cout, // 128 x Nout (f32 or bf16)
                   int K, int Ntot, int Nout)
{
  __shared__ __align__(16) unsigned short Abuf[2][TOK][LSTR];    // 2 x 128x64 bf16
  __shared__ __align__(16) unsigned short Bbuf[2][NTILE][LSTR];  // 2 x [n][k]

  const int tid   = threadIdx.x;
  const int lane  = tid & 31;
  const int wave  = tid >> 5;      // 0..7 : M tile index (16 rows each)
  const int hi    = lane >> 4;     // 0/1
  const int lo    = lane & 15;
  const int n0    = blockIdx.x * NTILE;
  const int qcols = Ntot >> 3;

  v8f acc[4] = {};

  issue_async_A(tid, A, K, 0, Abuf[0]);
  StageRegs R;
  load_global_tileB(R, tid, qw, qz, sc, Ntot, qcols, n0, 0);

  int p = 0;
  for (int kk = 0; kk < K; kk += KTILE, p ^= 1) {
    store_lds_tileB(R, tid, Bbuf[p]);
    if (kk + KTILE < K)
      load_global_tileB(R, tid, qw, qz, sc, Ntot, qcols, n0, kk + KTILE);
    wait_async0();                   // this wave's A(t) chunks landed in LDS
    __syncthreads();                 // publish A(t) + B(t) to all waves
    if (kk + KTILE < K)              // A(t+1) DMA overlaps the WMMA chain;
      issue_async_A(tid, A, K, kk + KTILE, Abuf[p ^ 1]);  // readers of p^1 done

    // ---- 2 k-subtiles x 4 n-subtiles = 8 WMMAs, loads hoisted per burst ----
    #pragma unroll
    for (int ks = 0; ks < 2; ++ks) {
      const int kb = ks << 5;          // k-subtile base (halfs)
      union { v16bf v; uint4 q[2]; } fa, fb[4];
      const unsigned short* arow = &Abuf[p][(wave << 4) + lo][kb];
      fa.q[0] = *(const uint4*)(arow + (hi << 3));        // K 0..7 | 8..15
      fa.q[1] = *(const uint4*)(arow + 16 + (hi << 3));   // K16..23|24..31
      #pragma unroll
      for (int nt = 0; nt < 4; ++nt) {
        const unsigned short* brow = &Bbuf[p][(nt << 4) + lo][kb];
        fb[nt].q[0] = *(const uint4*)(brow + (hi << 3));  // K 0..7 | 8..15
        fb[nt].q[1] = *(const uint4*)(brow + 16 + (hi << 3));
      }
      #pragma unroll
      for (int nt = 0; nt < 4; ++nt) {
        acc[nt] = __builtin_amdgcn_wmma_f32_16x16x32_bf16(
            false, fa.v, false, fb[nt].v, (short)0, acc[nt], false, false);
      }
    }
  }

  // ---- epilogue: C/D layout: VGPR v, lane -> (M = 16*wave + v + 8*hi, N) ----
  #pragma unroll
  for (int nt = 0; nt < 4; ++nt) {
    const int ncol = n0 + (nt << 4) + lo;
    #pragma unroll
    for (int v = 0; v < 8; ++v) {
      int m = (wave << 4) + (hi << 3) + v;
      if (CBF16) ((unsigned short*)Cout)[(size_t)m * Nout + ncol] = bf16_rna(acc[nt][v]);
      else       ((float*)Cout)[(size_t)m * Nout + ncol] = acc[nt][v];
    }
  }
}

// ---------------------------------------------------------------------------
// rmsnorm (+ optional fused residual add): bf16 normalized out, fp32 residual
// ---------------------------------------------------------------------------
__global__ __launch_bounds__(256)
void rmsnorm_add_kernel(const float* __restrict__ x,
                        const float* __restrict__ add,       // may be null
                        const float* __restrict__ w,
                        unsigned short* __restrict__ out_norm,  // bf16
                        float* __restrict__ out_resid,          // may be null
                        int Hd)
{
  __shared__ float red[256];
  const int row = blockIdx.x;
  const float* xr = x + (size_t)row * Hd;
  const float* ar = add ? add + (size_t)row * Hd : nullptr;

  float ss = 0.f;
  for (int c2 = threadIdx.x; c2 < (Hd >> 1); c2 += 256) {
    int c = c2 << 1;
    float v0 = xr[c]     + (ar ? ar[c]     : 0.f);
    float v1 = xr[c + 1] + (ar ? ar[c + 1] : 0.f);
    if (out_resid) {
      float2 rv; rv.x = v0; rv.y = v1;
      *(float2*)&out_resid[(size_t)row * Hd + c] = rv;
    }
    ss += v0 * v0 + v1 * v1;
  }
  red[threadIdx.x] = ss;
  __syncthreads();
  for (int s = 128; s > 0; s >>= 1) {
    if (threadIdx.x < s) red[threadIdx.x] += red[threadIdx.x + s];
    __syncthreads();
  }
  const float inv = rsqrtf(red[0] / (float)Hd + 1e-6f);
  for (int c2 = threadIdx.x; c2 < (Hd >> 1); c2 += 256) {
    int c = c2 << 1;
    float v0 = (xr[c]     + (ar ? ar[c]     : 0.f)) * inv * w[c];
    float v1 = (xr[c + 1] + (ar ? ar[c + 1] : 0.f)) * inv * w[c + 1];
    *(unsigned*)&out_norm[(size_t)row * Hd + c] = pack_bf16x2(v0, v1);
  }
}

__global__ __launch_bounds__(256)
void silu_mul_kernel(const float* __restrict__ gu,
                     unsigned short* __restrict__ act,   // bf16
                     int rows, int Icols)
{
  int p = blockIdx.x * 256 + threadIdx.x;          // pair index
  int half = Icols >> 1;
  if (p >= rows * half) return;
  int r = p / half, c = (p - r * half) << 1;
  float2 g = *(const float2*)&gu[(size_t)r * (2 * Icols) + c];
  float2 u = *(const float2*)&gu[(size_t)r * (2 * Icols) + Icols + c];
  float a0 = (g.x / (1.f + __expf(-g.x))) * u.x;
  float a1 = (g.y / (1.f + __expf(-g.y))) * u.y;
  *(unsigned*)&act[(size_t)r * Icols + c] = pack_bf16x2(a0, a1);
}

__global__ __launch_bounds__(256)
void add_kernel(const float* __restrict__ a, const float* __restrict__ b,
                float* __restrict__ out, int n)
{
  int idx = blockIdx.x * 256 + threadIdx.x;
  if (idx < n) out[idx] = a[idx] + b[idx];
}

// ---------------------------------------------------------------------------
extern "C" void kernel_launch(void* const* d_in, const int* in_sizes, int n_in,
                              void* d_out, int out_size, void* d_ws, size_t ws_size,
                              hipStream_t stream)
{
  (void)in_sizes; (void)n_in; (void)out_size; (void)ws_size;

  const float* x      = (const float*)d_in[0];
  const float* ln1_w  = (const float*)d_in[1];
  const float* ln2_w  = (const float*)d_in[2];
  const int*   qkv_qw = (const int*)  d_in[3];
  const int*   qkv_qz = (const int*)  d_in[4];
  const float* qkv_sc = (const float*)d_in[5];
  const int*   o_qw   = (const int*)  d_in[6];
  const int*   o_qz   = (const int*)  d_in[7];
  const float* o_sc   = (const float*)d_in[8];
  const int*   gu_qw  = (const int*)  d_in[9];
  const int*   gu_qz  = (const int*)  d_in[10];
  const float* gu_sc  = (const float*)d_in[11];
  const int*   dn_qw  = (const int*)  d_in[12];
  const int*   dn_qz  = (const int*)  d_in[13];
  const float* dn_sc  = (const float*)d_in[14];
  float* out = (float*)d_out;

  // workspace layout (bytes; every buffer size is 256B-aligned)
  char* wsp = (char*)d_ws;
  unsigned short* h1 = (unsigned short*)wsp; wsp += (size_t)TOK * HDIM * 2;  // bf16
  unsigned short* t1 = (unsigned short*)wsp; wsp += (size_t)TOK * HDIM * 2;  // bf16
  float* t2 = (float*)wsp;                   wsp += (size_t)TOK * HDIM * 4;  // f32
  float* r2 = (float*)wsp;                   wsp += (size_t)TOK * HDIM * 4;  // f32
  unsigned short* h2 = (unsigned short*)wsp; wsp += (size_t)TOK * HDIM * 2;  // bf16
  float* gu = (float*)wsp;                   wsp += (size_t)TOK * 2 * IDIM * 4;
  unsigned short* act = (unsigned short*)wsp; wsp += (size_t)TOK * IDIM * 2; // bf16
  float* t3 = (float*)wsp;

  // 1) h1 = rmsnorm(x, ln1)            (bf16)
  rmsnorm_add_kernel<<<TOK, 256, 0, stream>>>(x, nullptr, ln1_w, h1, nullptr, HDIM);

  // 2) t1 = h1 @ Wqkv[:, :H]           (bf16 out; only Q columns needed)
  gemm_awq_bf16<true><<<HDIM / NTILE, 256, 0, stream>>>(
      h1, qkv_qw, qkv_qz, qkv_sc, t1, HDIM, 3 * HDIM, HDIM);

  // 3) t2 = t1 @ Wo                    (f32 out)
  gemm_awq_bf16<false><<<HDIM / NTILE, 256, 0, stream>>>(
      t1, o_qw, o_qz, o_sc, t2, HDIM, HDIM, HDIM);

  // 4) r2 = x + t2 ; h2 = rmsnorm(r2, ln2)
  rmsnorm_add_kernel<<<TOK, 256, 0, stream>>>(x, t2, ln2_w, h2, r2, HDIM);

  // 5) gu = h2 @ Wgu                   (N = 2I, f32 out)
  gemm_awq_bf16<false><<<(2 * IDIM) / NTILE, 256, 0, stream>>>(
      h2, gu_qw, gu_qz, gu_sc, gu, HDIM, 2 * IDIM, 2 * IDIM);

  // 6) act = silu(gate) * up           (bf16)
  silu_mul_kernel<<<(TOK * (IDIM / 2) + 255) / 256, 256, 0, stream>>>(
      gu, act, TOK, IDIM);

  // 7) t3 = act @ Wdn                  (K = 11008, f32 out)
  gemm_awq_bf16<false><<<HDIM / NTILE, 256, 0, stream>>>(
      act, dn_qw, dn_qz, dn_sc, t3, IDIM, HDIM, HDIM);

  // 8) out = r2 + t3
  add_kernel<<<(TOK * HDIM + 255) / 256, 256, 0, stream>>>(r2, t3, out, TOK * HDIM);
}
